// Encoder_Baseline_28570122453316
// MI455X (gfx1250) — matrix operand order repaired and checked by reading, not compiled, not run on hardware
//
#include <hip/hip_runtime.h>

// ---------------------------------------------------------------------------
// GRU-GCC forward for MI455X (gfx1250, wave32, WMMA + TDM, double-buffered).
//
// GEMMs: V_WMMA_F32_16X16X32_BF16, block tile 64(M) x 128(N), 8 waves,
// each wave owns a 16x64 patch (4 accumulators; 4 WMMAs per K=32 chunk
// sharing one A fragment).  LDS is double-buffered: while chunk i is being
// consumed by WMMAs, chunk i+1 is staged (W tile via Tensor Data Mover,
// A tile via global_load_b128 + v_cvt_pk_bf16_f32), so TDM/global latency
// hides under the matrix ops and only one barrier per K-chunk is needed.
//
// W matrices are pre-converted to bf16 once; TDM descriptor uses pad_enable
// to reproduce the 80-byte padded LDS row stride.
// Fragment layouts follow CDNA5 ISA sec 7.12.2.
// Workspace requirement: ~189 MB (layout in kernel_launch).
// ---------------------------------------------------------------------------

#define BB 256
#define TT 64
#define DIN 2048
#define HH 512

typedef __bf16    v16bf __attribute__((ext_vector_type(16)));
typedef __bf16    v8bf  __attribute__((ext_vector_type(8)));
typedef float     v8f   __attribute__((ext_vector_type(8)));
typedef unsigned  v4u   __attribute__((ext_vector_type(4)));
typedef int       v8i   __attribute__((ext_vector_type(8)));
typedef int       v4i   __attribute__((ext_vector_type(4)));

struct ASpec {
  const float* p[4];       // per-segment base pointer (row 0 of segment)
  long long    stride[4];  // per-segment row stride (floats)
  int          segRows;    // rows per segment
};

__device__ __forceinline__ const float* aRowPtr(const ASpec& a, int r) {
  int seg = r / a.segRows;
  int rr  = r - seg * a.segRows;
  return a.p[seg] + (long long)rr * a.stride[seg];
}

__device__ __forceinline__ v8bf cvt8(float4 f0, float4 f1) {
  v8bf p = { (__bf16)f0.x, (__bf16)f0.y, (__bf16)f0.z, (__bf16)f0.w,
             (__bf16)f1.x, (__bf16)f1.y, (__bf16)f1.z, (__bf16)f1.w };
  return p;
}

// TDM: load a [rows x 32] bf16 tile (row stride K elements) from global into
// LDS at lds_byte_addr, padding LDS rows 32 elems (64B) + 16B -> 80B stride.
// D# layout per CDNA5 ISA sec 8.3/8.4.
__device__ __forceinline__ void tdm_load_w_tile(const __bf16* gsrc,
                                                unsigned lds_byte_addr,
                                                int K, int rows) {
  unsigned long long ga = (unsigned long long)(size_t)gsrc;
  v4u g0;
  g0[0] = 1u;                                        // count=1, user mode
  g0[1] = lds_byte_addr;                             // lds_addr
  g0[2] = (unsigned)(ga & 0xFFFFFFFFu);              // global_addr[31:0]
  g0[3] = (unsigned)((ga >> 32) & 0x1FFFFFFu) | (2u << 30);  // addr[56:32]|type=2
  v8i g1;
  // data_size=1(2B), pad_enable=1, pad_interval=3(16 dwords), pad_amount=3(4 dwords)
  g1[0] = (int)((1u << 16) | (1u << 20) | (3u << 22) | (3u << 25));
  g1[1] = (int)(((unsigned)K & 0xFFFFu) << 16);      // tensor_dim0[15:0]
  g1[2] = (int)((((unsigned)K >> 16) & 0xFFFFu) | (((unsigned)rows & 0xFFFFu) << 16));
  g1[3] = (int)(32u << 16);                          // tile_dim0 = 32
  g1[4] = (int)(unsigned)rows;                       // tile_dim1 = rows
  g1[5] = (int)(unsigned)K;                          // tensor_dim0_stride[31:0]
  g1[6] = 0;
  g1[7] = 0;
  v4i z4 = {0, 0, 0, 0};
#if defined(__clang_major__) && (__clang_major__ >= 23)
  v8i z8 = {0, 0, 0, 0, 0, 0, 0, 0};
  __builtin_amdgcn_tensor_load_to_lds(g0, g1, z4, z4, z8, 0);
#else
  __builtin_amdgcn_tensor_load_to_lds(g0, g1, z4, z4, 0);
#endif
}

// C[M,N] = A[M,K] * Wb[N,K]^T + bias[N]   (Wb is bf16, row-major)
// grid = (N/128, M/64), block = 256 threads (8 waves).
__global__ __launch_bounds__(256) void wmma_gemm_bf16(
    ASpec a, const __bf16* __restrict__ Wb, const float* __restrict__ bias,
    float* __restrict__ C, int K, int ldc)
{
  // row stride 40 bf16 = 80 B (multiple of 16 B -> aligned b128 LDS accesses)
  __shared__ __bf16 Atile[2][64][40];
  __shared__ __bf16 Wtile[2][128][40];

  const int tid  = threadIdx.x;
  const int wv   = tid >> 5;        // wave 0..7
  const int lane = tid & 31;
  const int half = lane >> 4;
  const int l16  = lane & 15;

  const int tileM = blockIdx.y * 64;
  const int tileN = blockIdx.x * 128;

  const int mrow = (wv & 3) * 16 + l16;   // A-frag row within tile
  const int nb   = (wv >> 2) * 64;        // 64-col patch base within tile

  const unsigned wtile_lds = (unsigned)(size_t)&Wtile[0][0][0];  // low 32 bits = LDS off
  const int arow_s = tid >> 2;            // A staging row for this thread
  const int acol_s = (tid & 3) << 3;      // A staging col (8 elems)

  v8f acc[4];
  #pragma unroll
  for (int cg = 0; cg < 4; ++cg) acc[cg] = (v8f){0.f,0.f,0.f,0.f,0.f,0.f,0.f,0.f};

  const int nIter = K >> 5;   // K/32 chunks

  // ---- prologue: stage chunk 0 into buffer 0
  if (tid < 32)
    tdm_load_w_tile(Wb + (long long)tileN * K, wtile_lds, K, 128);
  {
    const float* sa = aRowPtr(a, tileM + arow_s) + acol_s;
    float4 f0 = *(const float4*)sa;
    float4 f1 = *(const float4*)(sa + 4);
    *(v8bf*)&Atile[0][arow_s][acol_s] = cvt8(f0, f1);
    __builtin_prefetch(sa + 32, 0, 3);
  }
  if (tid < 32) __builtin_amdgcn_s_wait_tensorcnt(0);
  __syncthreads();

  for (int i = 0; i < nIter; ++i) {
    const int cur  = i & 1;
    const int more = (i + 1 < nIter);

    // ---- stage chunk i+1 into the other buffer (hidden under WMMAs)
    if (more) {
      int kk = (i + 1) << 5;
      if (tid < 32)
        tdm_load_w_tile(Wb + (long long)tileN * K + kk,
                        wtile_lds + (unsigned)((cur ^ 1) * 128 * 40 * 2), K, 128);
      const float* sa = aRowPtr(a, tileM + arow_s) + kk + acol_s;
      float4 f0 = *(const float4*)sa;
      float4 f1 = *(const float4*)(sa + 4);
      *(v8bf*)&Atile[cur ^ 1][arow_s][acol_s] = cvt8(f0, f1);
      __builtin_prefetch(sa + 32, 0, 3);   // speculative: chunk i+2
    }

    // ---- A fragment: two contiguous 16B runs per lane (shared by 4 WMMAs)
    const __bf16* ar = &Atile[cur][mrow][0];
    v8bf alo = *(const v8bf*)(ar + 8 * half);
    v8bf ahi = *(const v8bf*)(ar + 16 + 8 * half);
    v16bf afrag = __builtin_shufflevector(alo, ahi,
        0,1,2,3,4,5,6,7,8,9,10,11,12,13,14,15);

    // ---- 4 independent WMMAs (one per 16-col group of this wave's patch)
    #pragma unroll
    for (int cg = 0; cg < 4; ++cg) {
      const __bf16* br = &Wtile[cur][nb + cg * 16 + l16][0];
      v8bf blo = *(const v8bf*)(br + 8 * half);
      v8bf bhi = *(const v8bf*)(br + 16 + 8 * half);
      v16bf bfrag = __builtin_shufflevector(blo, bhi,
          0,1,2,3,4,5,6,7,8,9,10,11,12,13,14,15);
      acc[cg] = __builtin_amdgcn_wmma_f32_16x16x32_bf16(
          false, afrag, false, bfrag, (short)0, acc[cg], false, false);
    }

    // ---- single barrier per chunk: joins "next buffer written" with
    //      "current buffer consumed"
    if (more) {
      if (tid < 32) __builtin_amdgcn_s_wait_tensorcnt(0);
      __syncthreads();
    }
  }

  // ---- epilogue: C/D layout -> lane holds col l16, VGPR r holds row r+8*half
  const int mbase = tileM + (wv & 3) * 16;
  #pragma unroll
  for (int cg = 0; cg < 4; ++cg) {
    int n = tileN + nb + cg * 16 + l16;
    float bv = bias ? bias[n] : 0.f;
    #pragma unroll
    for (int r = 0; r < 8; ++r)
      C[(long long)(mbase + r + 8 * half) * ldc + n] = acc[cg][r] + bv;
  }
}

// Convert all weights to bf16; build concatenations:
//   Wihb = bf16(W_ih) [1536x2048]      Wfhb = bf16(W_fh0) [1024x2048]
//   Wcatb = bf16(W_hh||Wq) [2048x512]  Wkvb = bf16(Wk||Wv) [1024x512]
//   bcat = b_hh||bq (f32)              bkv = bk||bv (f32)
__global__ void prep_weights(const float* __restrict__ W_ih,
                             const float* __restrict__ W_fh0,
                             const float* __restrict__ W_hh, const float* __restrict__ b_hh,
                             const float* __restrict__ Wq,   const float* __restrict__ bq,
                             const float* __restrict__ Wk,   const float* __restrict__ bk,
                             const float* __restrict__ Wv,   const float* __restrict__ bv,
                             __bf16* __restrict__ Wihb, __bf16* __restrict__ Wfhb,
                             __bf16* __restrict__ Wcatb, __bf16* __restrict__ Wkvb,
                             float* __restrict__ bcat, float* __restrict__ bkv)
{
  long long i0 = (long long)blockIdx.x * blockDim.x + threadIdx.x;
  long long gs = (long long)gridDim.x * blockDim.x;
  for (long long i = i0; i < 1536LL * 2048; i += gs) Wihb[i] = (__bf16)W_ih[i];
  for (long long i = i0; i < 1024LL * 2048; i += gs) Wfhb[i] = (__bf16)W_fh0[i];
  for (long long i = i0; i < 2048LL * 512; i += gs)
    Wcatb[i] = (__bf16)((i < 1536LL * 512) ? W_hh[i] : Wq[i - 1536LL * 512]);
  for (long long i = i0; i < 1024LL * 512; i += gs)
    Wkvb[i] = (__bf16)((i < 512LL * 512) ? Wk[i] : Wv[i - 512LL * 512]);
  for (long long i = i0; i < 2048; i += gs) bcat[i] = (i < 1536) ? b_hh[i] : bq[i - 1536];
  for (long long i = i0; i < 1024; i += gs) bkv[i] = (i < 512) ? bk[i] : bv[i - 512];
}

__device__ __forceinline__ float sigmoidf_(float x) { return 1.f / (1.f + expf(-x)); }

// One block per batch row b. ghq[256,2048] = [gh_r|gh_i|gh_n|q].
// kv[1024,1024]: rows = [t1(256); i_r(256); t2(256); i_i(256)], cols = [k(512)|v(512)].
__global__ __launch_bounds__(256) void attn_update_step(
    const float* __restrict__ ghq, const float* __restrict__ kv,
    const float* __restrict__ gi_all, int t,
    const float* __restrict__ h_in, float* __restrict__ h_out,
    float* __restrict__ out, float* __restrict__ h_last, int isLast)
{
  const int b = blockIdx.x;
  const int j = threadIdx.x;                 // 256 threads
  const float scale = 0.04419417382415922f;  // 1/sqrt(512)

  __shared__ float red[4][256];

  const float* qrow = ghq + (long long)b * 2048 + 1536;
  float part0 = 0.f, part1 = 0.f, part2 = 0.f, part3 = 0.f;
  #pragma unroll
  for (int it = 0; it < 2; ++it) {
    int hh = j + it * 256;
    float qv = qrow[hh];
    part0 += qv * kv[(long long)(0 * 256 + b) * 1024 + hh];  // t1   (reset s0)
    part1 += qv * kv[(long long)(1 * 256 + b) * 1024 + hh];  // i_r  (reset s1)
    part2 += qv * kv[(long long)(2 * 256 + b) * 1024 + hh];  // t2   (input s0)
    part3 += qv * kv[(long long)(3 * 256 + b) * 1024 + hh];  // i_i  (input s1)
  }
  red[0][j] = part0; red[1][j] = part1; red[2][j] = part2; red[3][j] = part3;
  __syncthreads();
  for (int off = 128; off > 0; off >>= 1) {
    if (j < off) {
      red[0][j] += red[0][j + off];
      red[1][j] += red[1][j + off];
      red[2][j] += red[2][j + off];
      red[3][j] += red[3][j + off];
    }
    __syncthreads();
  }

  float sr0 = red[0][0] * scale, sr1 = red[1][0] * scale;
  float si0 = red[2][0] * scale, si1 = red[3][0] * scale;
  float mr = fmaxf(sr0, sr1), mi = fmaxf(si0, si1);
  float er0 = expf(sr0 - mr), er1 = expf(sr1 - mr);
  float ei0 = expf(si0 - mi), ei1 = expf(si1 - mi);
  float pr0 = er0 / (er0 + er1), pr1 = er1 / (er0 + er1);
  float pi0 = ei0 / (ei0 + ei1), pi1 = ei1 / (ei0 + ei1);

  const float* ghb = ghq + (long long)b * 2048;
  const float* gib = gi_all + ((long long)b * TT + t) * 1536;
  #pragma unroll
  for (int it = 0; it < 2; ++it) {
    int hh = j + it * 256;
    float ar = pr0 * kv[(long long)(0 * 256 + b) * 1024 + 512 + hh]
             + pr1 * kv[(long long)(1 * 256 + b) * 1024 + 512 + hh];
    float ai = pi0 * kv[(long long)(2 * 256 + b) * 1024 + 512 + hh]
             + pi1 * kv[(long long)(3 * 256 + b) * 1024 + 512 + hh];
    float rg = sigmoidf_(ar + ghb[hh]);
    float ig = sigmoidf_(ai + ghb[512 + hh]);
    float ng = tanhf(gib[1024 + hh] + rg * ghb[1024 + hh]);
    float hv = h_in[(long long)b * 512 + hh];
    float hy = ng + ig * (hv - ng);
    h_out[(long long)b * 512 + hh] = hy;
    out[((long long)b * TT + t) * 512 + hh] = hy;
    if (isLast) h_last[(long long)b * 512 + hh] = hy;
  }
}

static inline ASpec simpleA(const float* A, long long lda, int M) {
  ASpec a;
  for (int i = 0; i < 4; ++i) { a.p[i] = A; a.stride[i] = lda; }
  a.segRows = M;
  return a;
}

extern "C" void kernel_launch(void* const* d_in, const int* in_sizes, int n_in,
                              void* d_out, int out_size, void* d_ws, size_t ws_size,
                              hipStream_t stream) {
  (void)in_sizes; (void)n_in; (void)out_size; (void)ws_size;
  const float* input_feats = (const float*)d_in[0];   // [256,64,2048]
  const float* aux0        = (const float*)d_in[1];   // [256,64,2048]
  const float* W_ih        = (const float*)d_in[2];   // [1536,2048]
  const float* b_ih        = (const float*)d_in[3];   // [1536]
  const float* W_fh0       = (const float*)d_in[4];   // [1024,2048]
  const float* b_fh0       = (const float*)d_in[5];   // [1024]
  const float* W_hh        = (const float*)d_in[6];   // [1536,512]
  const float* b_hh        = (const float*)d_in[7];
  const float* Wq          = (const float*)d_in[8];   // [512,512]
  const float* bq          = (const float*)d_in[9];
  const float* Wk          = (const float*)d_in[10];
  const float* bk          = (const float*)d_in[11];
  const float* Wv          = (const float*)d_in[12];
  const float* bv          = (const float*)d_in[13];

  // Workspace layout: fp32 region then bf16 region (~189 MB total)
  float* ws      = (float*)d_ws;
  float* gi_all  = ws;                         // 16384*1536 = 25165824
  float* gf_all  = gi_all + 25165824LL;        // 16384*1024 = 16777216
  float* ghq     = gf_all + 16777216LL;        // 256*2048   =   524288
  float* kvbuf   = ghq    + 524288LL;          // 1024*1024  =  1048576
  float* h       = kvbuf  + 1048576LL;         // 256*512    =   131072
  float* bcat    = h      + 131072LL;          // 2048
  float* bkv     = bcat   + 2048LL;            // 1024
  __bf16* Wihb   = (__bf16*)(bkv + 1024LL);    // 1536*2048
  __bf16* Wfhb   = Wihb  + 3145728LL;          // 1024*2048
  __bf16* Wcatb  = Wfhb  + 2097152LL;          // 2048*512
  __bf16* Wkvb   = Wcatb + 1048576LL;          // 1024*512

  float* out    = (float*)d_out;               // [256,64,512]
  float* h_last = out + (long long)BB * TT * HH;

  hipMemsetAsync(h, 0, (size_t)BB * HH * sizeof(float), stream);
  prep_weights<<<dim3(2048), dim3(256), 0, stream>>>(
      W_ih, W_fh0, W_hh, b_hh, Wq, bq, Wk, bk, Wv, bv,
      Wihb, Wfhb, Wcatb, Wkvb, bcat, bkv);

  // gi_all[16384,1536] = input_feats @ W_ih^T + b_ih
  wmma_gemm_bf16<<<dim3(1536 / 128, 16384 / 64), 256, 0, stream>>>(
      simpleA(input_feats, DIN, BB * TT), Wihb, b_ih, gi_all, DIN, 1536);
  // gf_all[16384,1024] = aux0 @ W_fh0^T + b_fh0
  wmma_gemm_bf16<<<dim3(1024 / 128, 16384 / 64), 256, 0, stream>>>(
      simpleA(aux0, DIN, BB * TT), Wfhb, b_fh0, gf_all, DIN, 1024);

  for (int t = 0; t < TT; ++t) {
    // ghq[256,2048] = h @ [W_hh||Wq]^T + [b_hh||bq]
    wmma_gemm_bf16<<<dim3(2048 / 128, 256 / 64), 256, 0, stream>>>(
        simpleA(h, HH, BB), Wcatb, bcat, ghq, HH, 2048);

    // kv[1024,1024] = [t1; i_r; t2; i_i] @ [Wk||Wv]^T + [bk||bv]
    ASpec akv;
    akv.segRows   = BB;
    akv.p[0]      = gf_all + (long long)t * 1024;         // t1  = gf[:,t,0:512]
    akv.stride[0] = (long long)TT * 1024;
    akv.p[1]      = gi_all + (long long)t * 1536;         // i_r = gi[:,t,0:512]
    akv.stride[1] = (long long)TT * 1536;
    akv.p[2]      = gf_all + (long long)t * 1024 + 512;   // t2  = gf[:,t,512:1024]
    akv.stride[2] = (long long)TT * 1024;
    akv.p[3]      = gi_all + (long long)t * 1536 + 512;   // i_i = gi[:,t,512:1024]
    akv.stride[3] = (long long)TT * 1536;
    wmma_gemm_bf16<<<dim3(1024 / 128, 1024 / 64), 256, 0, stream>>>(
        akv, Wkvb, bkv, kvbuf, HH, 1024);

    attn_update_step<<<dim3(BB), dim3(256), 0, stream>>>(
        ghq, kvbuf, gi_all, t, h, h, out, h_last, (t == TT - 1) ? 1 : 0);
  }
}
